// Memory_global_43533788512939
// MI455X (gfx1250) — matrix operand order repaired and checked by reading, not compiled, not run on hardware
//
#include <hip/hip_runtime.h>

#define C_DIM 256
#define F_DIM 1000
#define F_PAD 1024
#define FT 32
#define NFT (F_PAD / FT)          // 32 f-tiles
#define ROWS_PER_WG 32            // 2 waves x 16 rows
#define THREADS 64
#define N_ROWS (32 * 64 * 64)     // 131072
#define ROWS_PER_BATCH 4096

typedef __attribute__((ext_vector_type(16))) __bf16 v16bf;
typedef __attribute__((ext_vector_type(8)))  float  v8f;

__device__ __forceinline__ v8f zero_v8f() {
  v8f z;
#pragma unroll
  for (int i = 0; i < 8; ++i) z[i] = 0.0f;
  return z;
}

// ---------------------------------------------------------------------------
// Kernel 1: weight fp32 [1000,256] -> bf16 in BOTH orientations, zero-padded
// to 1024 rows:  wb[f][c] (1024x256) and wbT[c][f] (256x1024), so every LDS
// tile stage in the main kernel is a pure byte-copy done with async DMA.
// Also zeroes the 32 per-batch entropy accumulators (needed every launch).
// ---------------------------------------------------------------------------
__global__ void weight_prep(const float* __restrict__ w,
                            __bf16* __restrict__ wb,
                            __bf16* __restrict__ wbT,
                            float* __restrict__ entacc) {
  int i = blockIdx.x * 256 + threadIdx.x;     // over F_PAD * C_DIM
  int f = i >> 8;
  int c = i & 255;
  float v = (f < F_DIM) ? w[i] : 0.0f;
  __bf16 h = (__bf16)v;
  wb[i] = h;
  wbT[c * F_PAD + f] = h;
  if (i < 32) entacc[i] = 0.0f;
}

// ---------------------------------------------------------------------------
// Kernel 2: fused  dist = X.W^T -> online softmax -> out = att.W  + entropy.
// Flash-attention structure with W playing both K and V.  2 waves/WG, each
// wave owns 16 rows x 256 cols.  Weight tiles stream L2->LDS with async DMA,
// ping-pong double buffered against ASYNCcnt so DMA overlaps WMMA compute.
// ---------------------------------------------------------------------------
__device__ __forceinline__ void stage_tile_async(int ft, unsigned ldsA, unsigned ldsT,
                                                 const __bf16* __restrict__ wb,
                                                 const __bf16* __restrict__ wbT,
                                                 int tid) {
  const __bf16* gA = wb  + ft * (FT * C_DIM);   // 16 KB contiguous
  const __bf16* gT = wbT + ft * FT;             // 256 rows of 64 B, stride 2 KB
#pragma unroll
  for (int j = 0; j < 16; ++j) {
    const int i4 = tid + THREADS * j;           // 1024 x 16-byte chunks
    const unsigned offA = (unsigned)(i4 * 16);
    asm volatile("global_load_async_to_lds_b128 %0, %1, %2"
                 :: "v"(ldsA + offA), "v"(offA), "s"(gA) : "memory");
    const int c = i4 >> 2, q = i4 & 3;
    const unsigned offT = (unsigned)(c * (F_PAD * 2) + q * 16);
    asm volatile("global_load_async_to_lds_b128 %0, %1, %2"
                 :: "v"(ldsT + offA), "v"(offT), "s"(gT) : "memory");
  }
}

__launch_bounds__(THREADS, 1)
__global__ void memory_fused(const float* __restrict__ x,
                             const __bf16* __restrict__ wb,
                             const __bf16* __restrict__ wbT,
                             float* __restrict__ out,
                             float* __restrict__ entacc) {
  __shared__ __align__(32) __bf16 wA[2][FT * C_DIM];  // [buf][f][c], 2 x 16 KB
  __shared__ __align__(32) __bf16 wT[2][C_DIM * FT];  // [buf][c][f], 2 x 16 KB
  __shared__ __align__(32) __bf16 xA[2][8][512];      // per-wave X A-frags, 16 KB
  __shared__ __align__(32) __bf16 pA[2][512];         // per-wave P A-frag slab

  const int tid    = threadIdx.x;
  const int wave   = tid >> 5;
  const int lane   = tid & 31;
  const int laneLo = lane & 15;
  const int laneHi = lane >> 4;

  const long rowBase = (long)blockIdx.x * ROWS_PER_WG + wave * 16;

  const unsigned ldsA0 = (unsigned)(uintptr_t)&wA[0][0];
  const unsigned ldsA1 = (unsigned)(uintptr_t)&wA[1][0];
  const unsigned ldsT0 = (unsigned)(uintptr_t)&wT[0][0];
  const unsigned ldsT1 = (unsigned)(uintptr_t)&wT[1][0];

  // ---- prologue: start DMA for tiles 0 and 1 ----
  stage_tile_async(0, ldsA0, ldsT0, wb, wbT, tid);
  stage_tile_async(1, ldsA1, ldsT1, wb, wbT, tid);

  // ---- stage this wave's 16 x-rows into LDS in A-fragment layout ----
  // A (16x32 bf16): lane la holds row M=la&15; halves j=0..15 -> K=16*laHi+j.
  // Lane-private: write & read back identity.
  {
    const float* xr = x + (rowBase + laneLo) * C_DIM;
#pragma unroll
    for (int kt = 0; kt < 8; ++kt) {
      const int k0 = kt * 32 + 16 * laneHi;
      v16bf t;
#pragma unroll
      for (int j = 0; j < 8; ++j) {
        t[j]     = (__bf16)xr[k0 + j];
        t[8 + j] = (__bf16)xr[k0 + 8 + j];
      }
      *(v16bf*)&xA[wave][kt][lane * 16] = t;
    }
  }

  v8f acc[16];                       // 16 column tiles x (16x16 f32 C/D frag)
#pragma unroll
  for (int ct = 0; ct < 16; ++ct) acc[ct] = zero_v8f();

  float m[8], l[8], sd[8];           // per-row running max / sum-e / sum-e*d
#pragma unroll
  for (int r = 0; r < 8; ++r) { m[r] = -__builtin_inff(); l[r] = 0.0f; sd[r] = 0.0f; }

  for (int ft = 0; ft < NFT; ++ft) {
    const int cur = ft & 1;
    // Async loads complete in order per wave: waiting until only the 32
    // in-flight loads of tile ft+1 remain guarantees tile ft is resident.
    if (ft + 1 < NFT) asm volatile("s_wait_asynccnt 0x20" ::: "memory");
    else              asm volatile("s_wait_asynccnt 0x0"  ::: "memory");
    __syncthreads();

    const __bf16* wAc = &wA[cur][0];
    const __bf16* wTc = &wT[cur][0];

    // ---- dist tile: 16 rows x 32 f  (two 16x16 chains over K = C = 256) ----
    v8f d0 = zero_v8f(), d1 = zero_v8f();
#pragma unroll
    for (int kt = 0; kt < 8; ++kt) {
      v16bf a  = *(const v16bf*)&xA[wave][kt][lane * 16];
      // B = W^T chunk: lane la needs w[f = s*16+laneLo][c = kt*32+16*laneHi .. +16)
      v16bf b0 = *(const v16bf*)&wAc[(laneLo)      * C_DIM + kt * 32 + 16 * laneHi];
      v16bf b1 = *(const v16bf*)&wAc[(16 + laneLo) * C_DIM + kt * 32 + 16 * laneHi];
      d0 = __builtin_amdgcn_wmma_f32_16x16x32_bf16(false, a, false, b0, (short)0, d0, false, false);
      d1 = __builtin_amdgcn_wmma_f32_16x16x32_bf16(false, a, false, b1, (short)0, d1, false, false);
    }

    // ---- online softmax + entropy stats + P scatter (C/D frag: row = r+8*laneHi,
    //      col = laneLo; row reductions are width-16 shuffles) ----
    const bool v0 = (ft * FT + laneLo      < F_DIM);
    const bool v1 = (ft * FT + 16 + laneLo < F_DIM);
#pragma unroll
    for (int r = 0; r < 8; ++r) {
      float dv0 = d0[r], dv1 = d1[r];
      float a0 = v0 ? dv0 : -__builtin_inff();
      float a1 = v1 ? dv1 : -__builtin_inff();
      float tmax = fmaxf(a0, a1);
#pragma unroll
      for (int s = 1; s < 16; s <<= 1) tmax = fmaxf(tmax, __shfl_xor(tmax, s, 16));
      const float mn    = fmaxf(m[r], tmax);
      const float scale = __expf(m[r] - mn);          // exp(-inf)=0 on first tile
      float e0 = v0 ? __expf(dv0 - mn) : 0.0f;
      float e1 = v1 ? __expf(dv1 - mn) : 0.0f;
      float es = e0 + e1;
      float ds = e0 * dv0 + e1 * dv1;                 // masked e==0, dv finite
#pragma unroll
      for (int s = 1; s < 16; s <<= 1) {
        es += __shfl_xor(es, s, 16);
        ds += __shfl_xor(ds, s, 16);
      }
      l[r]  = l[r]  * scale + es;
      sd[r] = sd[r] * scale + ds;
      m[r]  = mn;
#pragma unroll
      for (int ct = 0; ct < 16; ++ct) acc[ct][r] *= scale;   // rescale row r

      // scatter P(M,K) into per-wave A-fragment slab:
      // la = M + 16*(K>>4), pos = K&15
      const int M = r + 8 * laneHi;
      { const int K = laneLo;
        pA[wave][(M + 16 * (K >> 4)) * 16 + (K & 15)] = (__bf16)e0; }
      { const int K = 16 + laneLo;
        pA[wave][(M + 16 * (K >> 4)) * 16 + (K & 15)] = (__bf16)e1; }
    }
    // same-wave DS RAW: DS ops are in-order per wave; wait + compiler fence
    asm volatile("s_wait_dscnt 0x0" ::: "memory");
    const v16bf pa = *(const v16bf*)&pA[wave][lane * 16];

    // ---- out accumulation: acc[ct] += P(16x32) x W(32x16) ----
#pragma unroll
    for (int ct = 0; ct < 16; ++ct) {
      // B = W chunk: lane la needs w[f = 16*laneHi .. +16)][c = ct*16+laneLo]
      v16bf b = *(const v16bf*)&wTc[(ct * 16 + laneLo) * FT + 16 * laneHi];
      acc[ct] = __builtin_amdgcn_wmma_f32_16x16x32_bf16(false, pa, false, b, (short)0, acc[ct], false, false);
    }

    // ---- all waves done reading buf `cur`: refill it with tile ft+2 ----
    __syncthreads();
    if (ft + 2 < NFT)
      stage_tile_async(ft + 2, cur ? ldsA1 : ldsA0, cur ? ldsT1 : ldsT0, wb, wbT, tid);
  }

  // ---- epilogue: normalize, store out, accumulate per-batch entropy ----
#pragma unroll
  for (int r = 0; r < 8; ++r) {
    const float inv = 1.0f / l[r];
    const long row  = rowBase + r + 8 * laneHi;
    float* op = out + row * C_DIM;
#pragma unroll
    for (int ct = 0; ct < 16; ++ct) op[ct * 16 + laneLo] = acc[ct][r] * inv;
    if (laneLo == 0) {
      // -sum a log a = log(l) + m - (sum e*d)/l
      const float ent = __logf(l[r]) + m[r] - sd[r] * inv;
      atomicAdd(&entacc[(int)(row >> 12)], ent);   // row / 4096 -> batch
    }
  }
}

// ---------------------------------------------------------------------------
// Kernel 3: ent[b] = sum / (H*W*F)
// ---------------------------------------------------------------------------
__global__ void ent_finalize(const float* __restrict__ entacc,
                             float* __restrict__ entout) {
  if (threadIdx.x < 32)
    entout[threadIdx.x] =
        entacc[threadIdx.x] * (1.0f / ((float)ROWS_PER_BATCH * (float)F_DIM));
}

extern "C" void kernel_launch(void* const* d_in, const int* in_sizes, int n_in,
                              void* d_out, int out_size, void* d_ws, size_t ws_size,
                              hipStream_t stream) {
  const float* x = (const float*)d_in[0];     // [32,64,64,256] fp32
  const float* w = (const float*)d_in[1];     // [1000,256] fp32
  float* out    = (float*)d_out;              // [N,256] then [32] entropy
  float* entout = out + (long)N_ROWS * C_DIM;

  float*  entacc = (float*)d_ws;                                    // 32 f32
  __bf16* wb     = (__bf16*)((char*)d_ws + 256);                    // 512 KB
  __bf16* wbT    = (__bf16*)((char*)d_ws + 256 + F_PAD * C_DIM * 2);// 512 KB

  weight_prep<<<(F_PAD * C_DIM) / 256, 256, 0, stream>>>(w, wb, wbT, entacc);
  memory_fused<<<N_ROWS / ROWS_PER_WG, THREADS, 0, stream>>>(x, wb, wbT, out, entacc);
  ent_finalize<<<1, 32, 0, stream>>>(entacc, entout);
}
